// UVInstantNGP_31928786879034
// MI455X (gfx1250) — matrix operand order repaired and checked by reading, not compiled, not run on hardware
//
#include <hip/hip_runtime.h>

typedef __attribute__((ext_vector_type(16))) _Float16 v16h;
typedef __attribute__((ext_vector_type(8)))  float    v8f;

#define W_IMG  1024
#define H_IMG  1024
#define NPIX   (W_IMG * H_IMG)
#define NLEV   16
#define TSIZE  (1u << 19)
#define TMASK  (TSIZE - 1u)
#define HIDDEN 64
#define DIN    32
#define PRIME1 2654435761u

#define BLOCK          256
#define WAVES          8
#define TILES_PER_WAVE 8
// pixels per block = WAVES * TILES_PER_WAVE * 16 = 1024  ->  grid = 1024

__device__ __forceinline__ void wave_lds_sync() {
  // cross-lane RAW through LDS within one wave: LDS ops are in-order per
  // wave; just ensure completion + stop compiler reordering.
  asm volatile("s_wait_dscnt 0" ::: "memory");
}

__device__ __forceinline__ v8f wmma16(v16h a, v16h b, v8f c) {
  return __builtin_amdgcn_wmma_f32_16x16x32_f16(
      /*neg_a=*/false, a, /*neg_b=*/false, b,
      /*c_mod=*/(short)0, c, /*reuse_a=*/false, /*reuse_b=*/false);
}

// B fragment: 16 contiguous halfs starting at p16 (4B aligned in LDS)
__device__ __forceinline__ v16h load_bfrag(const _Float16* p16) {
  union { v16h h; unsigned u[8]; } f;
  const unsigned* p = (const unsigned*)p16;
#pragma unroll
  for (int i = 0; i < 8; ++i) f.u[i] = p[i];
  return f.h;
}

// A fragment from an LDS row: u32 groups [kb2 .. kb2+7] (contiguous)
// (contiguous K order: lanes 0-15 K=0-15; lanes 16-31 K=16-31)
__device__ __forceinline__ v16h load_afrag(const _Float16* row, int kb2) {
  union { v16h h; unsigned u[8]; } f;
  const unsigned* p = (const unsigned*)row;
#pragma unroll
  for (int i = 0; i < 4; ++i) {
    f.u[i]     = p[kb2 + i];
    f.u[4 + i] = p[kb2 + 4 + i];
  }
  return f.h;
}

__global__ __launch_bounds__(BLOCK)
void UVInstantNGP_ngp_wmma_kernel(
    const float* __restrict__ tables,
    const float* __restrict__ W1, const float* __restrict__ B1,
    const float* __restrict__ W2, const float* __restrict__ B2,
    const float* __restrict__ W3, const float* __restrict__ B3,
    float* __restrict__ out)
{
  __shared__ __align__(16) _Float16 w1t[HIDDEN * DIN];     // [n][k] = W1[k][n]
  __shared__ __align__(16) _Float16 w2t[HIDDEN * HIDDEN];  // [n][k] = W2[k][n]
  __shared__ __align__(16) _Float16 w3t[16 * HIDDEN];      // [n][k], n>=3 zero
  __shared__ float b1s[HIDDEN];
  __shared__ float b2s[HIDDEN];
  __shared__ float b3s[16];
  __shared__ __align__(16) _Float16 encs[WAVES][16 * DIN];     // per-wave A stage
  __shared__ __align__(16) _Float16 hs[WAVES][16 * HIDDEN];    // per-wave act stage

  const int tid = threadIdx.x;

  // ---- stage weights as f16, transposed to [n][k] for contiguous B frags ----
  for (int i = tid; i < HIDDEN * DIN; i += BLOCK) {
    int n = i >> 5, k = i & 31;
    w1t[i] = (_Float16)W1[k * HIDDEN + n];
  }
  for (int i = tid; i < HIDDEN * HIDDEN; i += BLOCK) {
    int n = i >> 6, k = i & 63;
    w2t[i] = (_Float16)W2[k * HIDDEN + n];
  }
  for (int i = tid; i < 16 * HIDDEN; i += BLOCK) {
    int n = i >> 6, k = i & 63;
    w3t[i] = (n < 3) ? (_Float16)W3[k * 3 + n] : (_Float16)0.0f;
  }
  if (tid < HIDDEN) { b1s[tid] = B1[tid]; b2s[tid] = B2[tid]; }
  if (tid < 16)     { b3s[tid] = (tid < 3) ? B3[tid] : 0.0f; }
  __syncthreads();

  const int wave = tid >> 5;
  const int lane = tid & 31;
  const int nl   = lane & 15;   // pixel-in-tile (A rows) / N column (B,C,D)
  const int grp  = lane >> 4;   // half-wave group
  _Float16* enc = encs[wave];
  _Float16* hb  = hs[wave];

  // per-level grid resolutions, floor(16 * 64^(l/15)); split so the unrolled
  // index is compile-time and the grp choice is a simple select
  const float RLO[8] = {16.f, 21.f, 27.f, 36.f, 48.f, 64.f, 84.f, 111.f};
  const float RHI[8] = {147.f, 194.f, 256.f, 337.f, 445.f, 588.f, 776.f, 1024.f};

  for (int it = 0; it < TILES_PER_WAVE; ++it) {
    const int tile  = (blockIdx.x * WAVES + wave) * TILES_PER_WAVE + it;
    const int nbase = tile * 16;

    // ================= hash encoding (lane: pixel nl, 8 levels) =============
    {
      const int   n = nbase + nl;
      const float u = (float)(n & (W_IMG - 1)) * (1.0f / W_IMG);
      const float v = (float)(n >> 10)         * (1.0f / H_IMG);
      unsigned* encrow = (unsigned*)(enc + nl * DIN);
#pragma unroll
      for (int li = 0; li < 8; ++li) {
        const int   l = grp * 8 + li;
        const float r = grp ? RHI[li] : RLO[li];
        float sx = u * r, sy = v * r;
        float fx0 = floorf(sx), fy0 = floorf(sy);
        float fx = sx - fx0, fy = sy - fy0;
        unsigned cx = (unsigned)(int)fx0, cy = (unsigned)(int)fy0;
        const float2* tab = ((const float2*)tables) + (size_t)l * TSIZE;
        unsigned hy0 = cy * PRIME1, hy1 = (cy + 1u) * PRIME1;
        unsigned i00 = ( cx       ^ hy0) & TMASK;
        unsigned i10 = ((cx + 1u) ^ hy0) & TMASK;
        unsigned i01 = ( cx       ^ hy1) & TMASK;
        unsigned i11 = ((cx + 1u) ^ hy1) & TMASK;
        float2 f00 = tab[i00], f10 = tab[i10], f01 = tab[i01], f11 = tab[i11];
        float w00 = (1.f - fx) * (1.f - fy), w10 = fx * (1.f - fy);
        float w01 = (1.f - fx) * fy,         w11 = fx * fy;
        float e0 = w00 * f00.x + w10 * f10.x + w01 * f01.x + w11 * f11.x;
        float e1 = w00 * f00.y + w10 * f10.y + w01 * f01.y + w11 * f11.y;
        union { _Float16 h; unsigned short s; } c0, c1;
        c0.h = (_Float16)e0; c1.h = (_Float16)e1;
        encrow[l] = (unsigned)c0.s | ((unsigned)c1.s << 16);
      }
    }
    wave_lds_sync();

    // ================= layer 1: 16x64 = A(16x32) x W1, 4 N-tiles ============
    const v16h aenc = load_afrag(enc + nl * DIN, grp * 8);
#pragma unroll
    for (int t = 0; t < 4; ++t) {
      const int ncol = t * 16 + nl;
      v16h b = load_bfrag(w1t + ncol * DIN + grp * 16);
      v8f  c = {};
      c = wmma16(aenc, b, c);
      const float bias = b1s[ncol];
#pragma unroll
      for (int r = 0; r < 8; ++r) {
        float h1 = fmaxf(c[r] + bias, 0.0f);          // D: lane=N, vgpr r=M
        hb[(grp * 8 + r) * HIDDEN + ncol] = (_Float16)h1;
      }
    }
    wave_lds_sync();

    // ================= layer 2: K=64 via two chained WMMAs, 4 N-tiles =======
    const v16h a20 = load_afrag(hb + nl * HIDDEN,  0 + grp * 8);
    const v16h a21 = load_afrag(hb + nl * HIDDEN, 16 + grp * 8);
    // all hb reads issued above; per-wave in-order LDS makes the stores below safe
#pragma unroll
    for (int t = 0; t < 4; ++t) {
      const int ncol = t * 16 + nl;
      v8f c = {};
      c = wmma16(a20, load_bfrag(w2t + ncol * HIDDEN +  0 + grp * 16), c);
      c = wmma16(a21, load_bfrag(w2t + ncol * HIDDEN + 32 + grp * 16), c);
      const float bias = b2s[ncol];
#pragma unroll
      for (int r = 0; r < 8; ++r) {
        float h2 = fmaxf(c[r] + bias, 0.0f);
        hb[(grp * 8 + r) * HIDDEN + ncol] = (_Float16)h2;
      }
    }
    wave_lds_sync();

    // ================= layer 3 (N padded to 16, 3 valid) + sigmoid ==========
    const v16h a30 = load_afrag(hb + nl * HIDDEN,  0 + grp * 8);
    const v16h a31 = load_afrag(hb + nl * HIDDEN, 16 + grp * 8);
    v8f c3 = {};
    c3 = wmma16(a30, load_bfrag(w3t + nl * HIDDEN +  0 + grp * 16), c3);
    c3 = wmma16(a31, load_bfrag(w3t + nl * HIDDEN + 32 + grp * 16), c3);
    const float bias3 = b3s[nl];
    if (nl < 3) {                   // divergence only after the last WMMA
#pragma unroll
      for (int r = 0; r < 8; ++r) {
        float z = c3[r] + bias3;
        float s = 1.0f / (1.0f + __expf(-z));
        out[(size_t)nl * NPIX + (size_t)(nbase + grp * 8 + r)] = s;
      }
    }
  }
}

extern "C" void kernel_launch(void* const* d_in, const int* in_sizes, int n_in,
                              void* d_out, int out_size, void* d_ws, size_t ws_size,
                              hipStream_t stream) {
  const float* tables = (const float*)d_in[0];
  const float* W1 = (const float*)d_in[1];
  const float* b1 = (const float*)d_in[2];
  const float* W2 = (const float*)d_in[3];
  const float* b2 = (const float*)d_in[4];
  const float* W3 = (const float*)d_in[5];
  const float* b3 = (const float*)d_in[6];
  float* out = (float*)d_out;

  const int grid = NPIX / (WAVES * TILES_PER_WAVE * 16);   // 1024 blocks
  UVInstantNGP_ngp_wmma_kernel<<<grid, BLOCK, 0, stream>>>(
      tables, W1, b1, W2, b2, W3, b3, out);
}
